// GCNNet_58454504898644
// MI455X (gfx1250) — matrix operand order repaired and checked by reading, not compiled, not run on hardware
//
#include <hip/hip_runtime.h>
#include <hip/hip_bf16.h>

typedef _Float16 v16h __attribute__((ext_vector_type(16)));
typedef float    v8f  __attribute__((ext_vector_type(8)));
typedef float    v4f  __attribute__((ext_vector_type(4)));

#define F_IN 256
#define HID  64
#define NC   16

// ---------------- degree / normalization ----------------
__global__ void k_deg_init(float* __restrict__ deg, int n) {
  int i = blockIdx.x * blockDim.x + threadIdx.x;
  if (i < n) deg[i] = 1.0f;                 // self-loop contribution
}

__global__ void k_deg_count(const long long* __restrict__ dst,
                            float* __restrict__ deg, int E) {
  int e = blockIdx.x * blockDim.x + threadIdx.x;
  if (e < E) atomicAdd(&deg[(int)dst[e]], 1.0f);
}

__global__ void k_rsqrt(float* __restrict__ d, int n) {
  int i = blockIdx.x * blockDim.x + threadIdx.x;
  if (i < n) d[i] = rsqrtf(d[i]);           // deg >= 1 always (self-loops)
}

// ---------------- WMMA GEMM1: xw = f16(x) @ f16(W1), f32 accum ----------------
// per-wave: one 16-row M tile, all 64 N columns, K = 256 in 8 chunks of 32
__global__ __launch_bounds__(256) void k_gemm1(const float* __restrict__ x,
                                               const float* __restrict__ W1,
                                               float* __restrict__ xw,
                                               int mtiles) {
  __shared__ alignas(32) _Float16 w1s[HID * F_IN];   // transposed: [n][k]
  int tid = threadIdx.x;
  for (int i = tid; i < F_IN * HID; i += 256) {      // W1 row-major [k][n]
    int k = i >> 6, n = i & 63;
    w1s[n * F_IN + k] = (_Float16)W1[i];
  }
  __syncthreads();

  int wave = tid >> 5, lane = tid & 31;
  int mt = blockIdx.x * 8 + wave;
  if (mt >= mtiles) return;                          // wave-uniform

  int m0   = mt * 16;
  int lrow = lane & 15;       // M row (A) / N col (B,D)
  int kgrp = lane >> 4;       // lane-group K split
  const float* xr = x + (size_t)(m0 + lrow) * F_IN;

  v8f c0 = {}, c1 = {}, c2 = {}, c3 = {};
#pragma unroll
  for (int kc = 0; kc < F_IN / 32; ++kc) {
    int k0 = kc * 32;
    // A fragment 16x32 f16: halves 0..15 -> K = k0+kgrp*16+0..15
    v4f q0 = *(const v4f*)(xr + k0 + kgrp * 16);
    v4f q1 = *(const v4f*)(xr + k0 + kgrp * 16 + 4);
    v4f q2 = *(const v4f*)(xr + k0 + kgrp * 16 + 8);
    v4f q3 = *(const v4f*)(xr + k0 + kgrp * 16 + 12);
    v16h a;
#pragma unroll
    for (int j = 0; j < 4; ++j) {
      a[j]      = (_Float16)q0[j];
      a[4 + j]  = (_Float16)q1[j];
      a[8 + j]  = (_Float16)q2[j];
      a[12 + j] = (_Float16)q3[j];
    }
    // B fragments 32x16: lane group 0 holds K=k0..k0+15, group 1 K=k0+16..k0+31
    int kbase = k0 + kgrp * 16;
    v16h b0 = *(const v16h*)&w1s[(0  + lrow) * F_IN + kbase];
    v16h b1 = *(const v16h*)&w1s[(16 + lrow) * F_IN + kbase];
    v16h b2 = *(const v16h*)&w1s[(32 + lrow) * F_IN + kbase];
    v16h b3 = *(const v16h*)&w1s[(48 + lrow) * F_IN + kbase];
    c0 = __builtin_amdgcn_wmma_f32_16x16x32_f16(false, a, false, b0, (short)0, c0, false, false);
    c1 = __builtin_amdgcn_wmma_f32_16x16x32_f16(false, a, false, b1, (short)0, c1, false, false);
    c2 = __builtin_amdgcn_wmma_f32_16x16x32_f16(false, a, false, b2, (short)0, c2, false, false);
    c3 = __builtin_amdgcn_wmma_f32_16x16x32_f16(false, a, false, b3, (short)0, c3, false, false);
  }
  // D layout: VGPR r -> M = r + 8*kgrp, N = lane&15
#pragma unroll
  for (int r = 0; r < 8; ++r) {
    int row = m0 + r + kgrp * 8;
    float* orow = xw + (size_t)row * HID;
    orow[0  + lrow] = c0[r];
    orow[16 + lrow] = c1[r];
    orow[32 + lrow] = c2[r];
    orow[48 + lrow] = c3[r];
  }
}

// ---------------- layer-1 aggregation ----------------
// self-loop term: h1 = xw * dinv[i]^2   (t indexes float4 groups, 16 per node)
__global__ void k_self1(const float* __restrict__ xw, const float* __restrict__ dinv,
                        float* __restrict__ h1, int nvec) {
  int t = blockIdx.x * blockDim.x + threadIdx.x;
  if (t >= nvec) return;
  int node = t >> 4;
  float sn = dinv[node] * dinv[node];
  ((v4f*)h1)[t] = ((const v4f*)xw)[t] * sn;
}

// edge scatter: h1[dst] += xw[src] * dinv[src]*dinv[dst]  (thread = edge x 4-float group)
__global__ void k_scatter1(const long long* __restrict__ src, const long long* __restrict__ dst,
                           const float* __restrict__ dinv, const float* __restrict__ xw,
                           float* __restrict__ h1, int nwork) {
  int t = blockIdx.x * blockDim.x + threadIdx.x;
  if (t >= nwork) return;
  int e = t >> 4, g = t & 15;
  int s = (int)src[e], d = (int)dst[e];
  float nrm = dinv[s] * dinv[d];
  v4f m = ((const v4f*)xw)[s * 16 + g] * nrm;
  float* o = h1 + (size_t)d * HID + g * 4;
  atomicAdd(o + 0, m[0]);
  atomicAdd(o + 1, m[1]);
  atomicAdd(o + 2, m[2]);
  atomicAdd(o + 3, m[3]);
}

__global__ void k_bias_relu(float* __restrict__ h1, const float* __restrict__ b1, int nvec) {
  int t = blockIdx.x * blockDim.x + threadIdx.x;
  if (t >= nvec) return;
  int g = t & 15;
  v4f v = ((const v4f*)h1)[t] + ((const v4f*)b1)[g];
#pragma unroll
  for (int j = 0; j < 4; ++j) v[j] = fmaxf(v[j], 0.0f);
  ((v4f*)h1)[t] = v;
}

// ---------------- WMMA GEMM2: h2 = f16(h1) @ f16(W2) ----------------
__global__ __launch_bounds__(256) void k_gemm2(const float* __restrict__ h1,
                                               const float* __restrict__ W2,
                                               float* __restrict__ h2,
                                               int mtiles) {
  __shared__ alignas(32) _Float16 w2s[NC * HID];     // transposed: [n][k]
  int tid = threadIdx.x;
  for (int i = tid; i < HID * NC; i += 256) {        // W2 row-major [k][n]
    int k = i >> 4, n = i & 15;
    w2s[n * HID + k] = (_Float16)W2[i];
  }
  __syncthreads();

  int wave = tid >> 5, lane = tid & 31;
  int mt = blockIdx.x * 8 + wave;
  if (mt >= mtiles) return;

  int m0 = mt * 16, lrow = lane & 15, kgrp = lane >> 4;
  const float* hr = h1 + (size_t)(m0 + lrow) * HID;

  v8f c = {};
#pragma unroll
  for (int kc = 0; kc < 2; ++kc) {
    int k0 = kc * 32;
    v4f q0 = *(const v4f*)(hr + k0 + kgrp * 16);
    v4f q1 = *(const v4f*)(hr + k0 + kgrp * 16 + 4);
    v4f q2 = *(const v4f*)(hr + k0 + kgrp * 16 + 8);
    v4f q3 = *(const v4f*)(hr + k0 + kgrp * 16 + 12);
    v16h a;
#pragma unroll
    for (int j = 0; j < 4; ++j) {
      a[j]      = (_Float16)q0[j];
      a[4 + j]  = (_Float16)q1[j];
      a[8 + j]  = (_Float16)q2[j];
      a[12 + j] = (_Float16)q3[j];
    }
    v16h b = *(const v16h*)&w2s[lrow * HID + k0 + kgrp * 16];
    c = __builtin_amdgcn_wmma_f32_16x16x32_f16(false, a, false, b, (short)0, c, false, false);
  }
#pragma unroll
  for (int r = 0; r < 8; ++r)
    h2[(size_t)(m0 + r + kgrp * 8) * NC + lrow] = c[r];
}

// ---------------- layer-2 aggregation ----------------
__global__ void k_self2(const float* __restrict__ h2, const float* __restrict__ dinv,
                        const float* __restrict__ b2, float* __restrict__ out, int nvec) {
  int t = blockIdx.x * blockDim.x + threadIdx.x;
  if (t >= nvec) return;
  int node = t >> 2, g = t & 3;
  float sn = dinv[node] * dinv[node];
  ((v4f*)out)[t] = ((const v4f*)h2)[t] * sn + ((const v4f*)b2)[g];
}

__global__ void k_scatter2(const long long* __restrict__ src, const long long* __restrict__ dst,
                           const float* __restrict__ dinv, const float* __restrict__ h2,
                           float* __restrict__ out, int nwork) {
  int t = blockIdx.x * blockDim.x + threadIdx.x;
  if (t >= nwork) return;
  int e = t >> 2, g = t & 3;
  int s = (int)src[e], d = (int)dst[e];
  float nrm = dinv[s] * dinv[d];
  v4f m = ((const v4f*)h2)[s * 4 + g] * nrm;
  float* o = out + (size_t)d * NC + g * 4;
  atomicAdd(o + 0, m[0]);
  atomicAdd(o + 1, m[1]);
  atomicAdd(o + 2, m[2]);
  atomicAdd(o + 3, m[3]);
}

__global__ void k_logsoftmax(float* __restrict__ out, int n) {
  int i = blockIdx.x * blockDim.x + threadIdx.x;
  if (i >= n) return;
  float* r = out + (size_t)i * NC;
  float v[NC];
  float m = -3.4e38f;
#pragma unroll
  for (int j = 0; j < NC; ++j) { v[j] = r[j]; m = fmaxf(m, v[j]); }
  float s = 0.0f;
#pragma unroll
  for (int j = 0; j < NC; ++j) s += __expf(v[j] - m);
  float ls = m + __logf(s);
#pragma unroll
  for (int j = 0; j < NC; ++j) r[j] = v[j] - ls;
}

// ---------------- launch ----------------
extern "C" void kernel_launch(void* const* d_in, const int* in_sizes, int n_in,
                              void* d_out, int out_size, void* d_ws, size_t ws_size,
                              hipStream_t stream) {
  (void)n_in; (void)out_size; (void)ws_size;
  const float*     x   = (const float*)d_in[0];
  const float*     W1  = (const float*)d_in[1];
  const float*     b1  = (const float*)d_in[2];
  const float*     W2  = (const float*)d_in[3];
  const float*     b2  = (const float*)d_in[4];
  const long long* ei  = (const long long*)d_in[5];   // int64 [2, E]

  const int N = in_sizes[0] / F_IN;
  const int E = in_sizes[5] / 2;
  const long long* srcv = ei;
  const long long* dstv = ei + E;

  char* ws = (char*)d_ws;
  size_t off = 0;
  auto wsalloc = [&](size_t bytes) -> void* {
    void* p = ws + off;
    off = (off + bytes + 255) & ~(size_t)255;
    return p;
  };
  float* dinv = (float*)wsalloc((size_t)N * 4);            // deg -> rsqrt in place
  float* xw   = (float*)wsalloc((size_t)N * HID * 4);      // x@W1
  float* h1   = (float*)wsalloc((size_t)N * HID * 4);      // aggregated layer 1
  float* h2   = (float*)wsalloc((size_t)N * NC * 4);       // relu(h1)@W2
  float* out  = (float*)d_out;                             // aggregated layer 2 + log_softmax

  const int T = 256;
  auto blk = [](long long n, int t) { return (int)((n + t - 1) / t); };

  // normalization
  k_deg_init <<<blk(N, T), T, 0, stream>>>(dinv, N);
  k_deg_count<<<blk(E, T), T, 0, stream>>>(dstv, dinv, E);
  k_rsqrt    <<<blk(N, T), T, 0, stream>>>(dinv, N);

  // layer 1
  const int mtiles = N / 16;
  k_gemm1    <<<blk(mtiles, 8), T, 0, stream>>>(x, W1, xw, mtiles);
  k_self1    <<<blk((long long)N * 16, T), T, 0, stream>>>(xw, dinv, h1, N * 16);
  k_scatter1 <<<blk((long long)E * 16, T), T, 0, stream>>>(srcv, dstv, dinv, xw, h1, E * 16);
  k_bias_relu<<<blk((long long)N * 16, T), T, 0, stream>>>(h1, b1, N * 16);

  // layer 2
  k_gemm2    <<<blk(mtiles, 8), T, 0, stream>>>(h1, W2, h2, mtiles);
  k_self2    <<<blk((long long)N * 4, T), T, 0, stream>>>(h2, dinv, b2, out, N * 4);
  k_scatter2 <<<blk((long long)E * 4, T), T, 0, stream>>>(srcv, dstv, dinv, h2, out, E * 4);
  k_logsoftmax<<<blk(N, T), T, 0, stream>>>(out, N);
}
